// GAE_30571577213220
// MI455X (gfx1250) — compile-verified, matrix operand order checked by reading, not run on hardware
//
#include <hip/hip_runtime.h>

#define NATTRS   200
#define NOBJS    400
#define N_NODES  600
#define NPAIRS   (NATTRS * NOBJS)   // 80000
#define NEDGES   30000

typedef __attribute__((ext_vector_type(16))) __bf16 v16bf;
typedef __attribute__((ext_vector_type(8)))  float  v8f;

// ---------- bf16 helpers (RNE) ----------
__device__ __forceinline__ __bf16 f2bf(float f) {
  unsigned u = __builtin_bit_cast(unsigned, f);
  u += 0x7fffu + ((u >> 16) & 1u);
  unsigned short h = (unsigned short)(u >> 16);
  return __builtin_bit_cast(__bf16, h);
}
__device__ __forceinline__ float bf2f(__bf16 b) {
  unsigned short h = __builtin_bit_cast(unsigned short, b);
  unsigned u = ((unsigned)h) << 16;
  return __builtin_bit_cast(float, u);
}

// ---------- gfx1250 async global->LDS copy (16B per lane, ASYNCcnt tracked) ----------
__device__ __forceinline__ unsigned ldsoff(const void* p) {
  // flat LDS pointer = {shared_aperture[63:32], lds_byte_offset[31:0]}
  return (unsigned)(uintptr_t)p;
}
__device__ __forceinline__ void async_b128(unsigned lds_dst, const void* gsrc) {
  asm volatile("global_load_async_to_lds_b128 %0, %1, off"
               :: "v"(lds_dst), "v"(gsrc)
               : "memory");
}
__device__ __forceinline__ void wait_async0() {
  asm volatile("s_wait_asynccnt 0x0" ::: "memory");
}

// ---------- WMMA tile compute: 64x128 block tile, wave = 16(M) x 64(N) ----------
// sA : [64][32] bf16 row-major (M x K tile)
// sBT: [128][32] bf16, sBT[n][k] = B[k][n]  (contiguous 32B fragments per lane)
__device__ __forceinline__ void wmma_compute(const __bf16* sA, const __bf16* sBT,
                                             v8f acc[4], int lane, int wm, int wn) {
  const int lr = lane & 15;            // A: row in 16-strip / B: col in 16-strip
  const int kh = (lane >> 4) << 4;     // lanes 16..31 hold K=16..31 (contiguous k order, same for A and B)
  v16bf a = *reinterpret_cast<const v16bf*>(sA + (wm * 16 + lr) * 32 + kh);
#pragma unroll
  for (int ns = 0; ns < 4; ++ns) {
    v16bf b = *reinterpret_cast<const v16bf*>(sBT + (wn * 64 + ns * 16 + lr) * 32 + kh);
    acc[ns] = __builtin_amdgcn_wmma_f32_16x16x32_bf16(false, a, false, b,
                                                      (short)0, acc[ns], false, false);
  }
}

// ---------- synchronous f32->bf16 staging (small GEMMs) ----------
__device__ __forceinline__ void stage_A_f32(__bf16* sA, const float* __restrict__ A,
                                            int m0, int k0, int M, int K, int tid) {
#pragma unroll
  for (int i = 0; i < 2; ++i) {
    int lin = tid + i * 256;          // 512 float4 = 64 rows x 8 chunks
    int row = lin >> 3;
    int kq  = (lin & 7) << 2;
    int gm = m0 + row, gk = k0 + kq;
    float4 v = make_float4(0.f, 0.f, 0.f, 0.f);
    if (gm < M && gk < K)             // K always a multiple of 4 here
      v = *reinterpret_cast<const float4*>(A + (size_t)gm * K + gk);
    __bf16* d = sA + row * 32 + kq;
    d[0] = f2bf(v.x); d[1] = f2bf(v.y); d[2] = f2bf(v.z); d[3] = f2bf(v.w);
  }
}

__device__ __forceinline__ void stage_B_f32(__bf16* sBT, const float* __restrict__ W,
                                            int n0, int k0, int N, int K, int tid) {
#pragma unroll
  for (int i = 0; i < 4; ++i) {
    int lin = tid + i * 256;          // 1024 float4 = 32 k-rows x 32 chunks of N
    int k  = lin >> 5;
    int nq = (lin & 31) << 2;
    int gk = k0 + k, gn = n0 + nq;
    float4 v = make_float4(0.f, 0.f, 0.f, 0.f);
    if (gk < K && gn < N)             // N always a multiple of 4 here
      v = *reinterpret_cast<const float4*>(W + (size_t)gk * N + gn);
    sBT[(nq + 0) * 32 + k] = f2bf(v.x);
    sBT[(nq + 1) * 32 + k] = f2bf(v.y);
    sBT[(nq + 2) * 32 + k] = f2bf(v.z);
    sBT[(nq + 3) * 32 + k] = f2bf(v.w);
  }
}

// ---------- generic GEMM: C = act(A1@W1 [+ A2@W2] + bias), f32 in/out ----------
__global__ __launch_bounds__(256) void k_gemm_dual(
    const float* __restrict__ A1, const float* __restrict__ W1,
    const float* __restrict__ A2, const float* __restrict__ W2,
    const float* __restrict__ bias, float* __restrict__ C,
    int M, int N, int K, int relu) {
  __shared__ __align__(32) __bf16 sA[64 * 32];
  __shared__ __align__(32) __bf16 sBT[128 * 32];
  const int tid = threadIdx.x, lane = tid & 31, wid = tid >> 5;
  const int wm = wid & 3, wn = wid >> 2;
  const int m0 = blockIdx.y * 64, n0 = blockIdx.x * 128;
  v8f acc[4] = {};
  const int nsteps = (K + 31) >> 5;
  for (int pass = 0; pass < 2; ++pass) {
    const float* A = pass ? A2 : A1;
    const float* W = pass ? W2 : W1;
    if (A == nullptr) break;
    for (int s = 0; s < nsteps; ++s) {
      int k0 = s << 5;
      stage_A_f32(sA, A, m0, k0, M, K, tid);
      stage_B_f32(sBT, W, n0, k0, N, K, tid);
      __syncthreads();
      wmma_compute(sA, sBT, acc, lane, wm, wn);
      __syncthreads();
    }
  }
#pragma unroll
  for (int ns = 0; ns < 4; ++ns) {
    int gn = n0 + wn * 64 + ns * 16 + (lane & 15);
    if (gn >= N) continue;
    float bv = bias ? bias[gn] : 0.f;
#pragma unroll
    for (int r = 0; r < 8; ++r) {
      int gm = m0 + wm * 16 + ((lane >> 4) << 3) + r;   // C layout: lanes16-31 -> M+8
      if (gm < M) {
        float v = acc[ns][r] + bv;
        if (relu) v = fmaxf(v, 0.f);
        C[(size_t)gm * N + gn] = v;
      }
    }
  }
}

// ---------- pair GEMM: PE[p] = relu(U[i]+V[j]+bp1) @ Wp2 + bp2  (bf16 out) ----------
// WT = Wp2 transposed/converted: bf16 [896][1024], zero-padded (rows>=800, k>=1000).
// Double-buffered; B tiles staged with async global->LDS copies.
__global__ __launch_bounds__(256) void k_gemm_pair(
    const float* __restrict__ U, const float* __restrict__ V,
    const float* __restrict__ bp1, const __bf16* __restrict__ WT,
    const float* __restrict__ bp2, __bf16* __restrict__ PE) {
  const int K = 1000, KP = 1024, N = 800;
  __shared__ __align__(32) __bf16 sA[2][64 * 32];
  __shared__ __align__(32) __bf16 sBT[2][128 * 32];
  const int tid = threadIdx.x, lane = tid & 31, wid = tid >> 5;
  const int wm = wid & 3, wn = wid >> 2;
  const int m0 = blockIdx.y * 64, n0 = blockIdx.x * 128;
  v8f acc[4] = {};
  const int nsteps = KP >> 5;                // 32 (zero-padded K)

  auto stage = [&](int s, int buf) {
    const int k0 = s << 5;
    // B: async row copy from WT (padded -> no guards). 8KB = 2 x 256 lanes x 16B.
#pragma unroll
    for (int i = 0; i < 2; ++i) {
      int lin = tid + i * 256;
      int n = lin >> 2;
      int kq = (lin & 3) << 3;
      async_b128(ldsoff(&sBT[buf][n * 32 + kq]),
                 WT + (size_t)(n0 + n) * KP + k0 + kq);
    }
    // A (virtual): relu(U[p/400] + V[p%400] + bp1) -> bf16, VALU path.
#pragma unroll
    for (int i = 0; i < 2; ++i) {
      int lin = tid + i * 256;
      int row = lin >> 3;
      int kq  = (lin & 7) << 2;
      int p  = m0 + row;                     // always < 80000 (80000 % 64 == 0)
      int gk = k0 + kq;
      float4 v = make_float4(0.f, 0.f, 0.f, 0.f);
      if (gk < K) {
        int ai = p / NOBJS;
        int oj = p - ai * NOBJS;
        float4 u4 = *reinterpret_cast<const float4*>(U + (size_t)ai * K + gk);
        float4 o4 = *reinterpret_cast<const float4*>(V + (size_t)oj * K + gk);
        float4 b4 = *reinterpret_cast<const float4*>(bp1 + gk);
        v.x = fmaxf(u4.x + o4.x + b4.x, 0.f);
        v.y = fmaxf(u4.y + o4.y + b4.y, 0.f);
        v.z = fmaxf(u4.z + o4.z + b4.z, 0.f);
        v.w = fmaxf(u4.w + o4.w + b4.w, 0.f);
      }
      __bf16* d = &sA[buf][row * 32 + kq];
      d[0] = f2bf(v.x); d[1] = f2bf(v.y); d[2] = f2bf(v.z); d[3] = f2bf(v.w);
    }
  };

  stage(0, 0);
  wait_async0();
  __syncthreads();
  for (int s = 0; s < nsteps; ++s) {
    int cur = s & 1;
    if (s + 1 < nsteps) stage(s + 1, cur ^ 1);   // overlaps with WMMAs below
    wmma_compute(sA[cur], sBT[cur], acc, lane, wm, wn);
    wait_async0();
    __syncthreads();
  }
#pragma unroll
  for (int ns = 0; ns < 4; ++ns) {
    int gn = n0 + wn * 64 + ns * 16 + (lane & 15);
    if (gn >= N) continue;
    float bv = bp2[gn];
#pragma unroll
    for (int r = 0; r < 8; ++r) {
      int gm = m0 + wm * 16 + ((lane >> 4) << 3) + r;
      PE[(size_t)gm * N + gn] = f2bf(acc[ns][r] + bv);
    }
  }
}

// ---------- final GEMM: C = AB @ BT^T, AB bf16 [512,800], BT bf16 [80000,800] ----------
// Both operands staged with async global->LDS copies, double-buffered.
__global__ __launch_bounds__(256) void k_gemm_abt(
    const __bf16* __restrict__ AB, const __bf16* __restrict__ BT,
    float* __restrict__ C, int M, int N, int K) {
  __shared__ __align__(32) __bf16 sA[2][64 * 32];
  __shared__ __align__(32) __bf16 sBT[2][128 * 32];
  const int tid = threadIdx.x, lane = tid & 31, wid = tid >> 5;
  const int wm = wid & 3, wn = wid >> 2;
  const int m0 = blockIdx.y * 64, n0 = blockIdx.x * 128;
  v8f acc[4] = {};
  const int nsteps = K >> 5;                 // K=800 -> 25, exact; M,N tile-exact

  auto stage = [&](int s, int buf) {
    const int k0 = s << 5;
    {
      int row = tid >> 2;                    // A: 4KB = 256 lanes x 16B
      int kq  = (tid & 3) << 3;
      async_b128(ldsoff(&sA[buf][row * 32 + kq]),
                 AB + (size_t)(m0 + row) * K + k0 + kq);
    }
#pragma unroll
    for (int i = 0; i < 2; ++i) {            // B: 8KB = 2 x 256 lanes x 16B
      int lin = tid + i * 256;
      int n = lin >> 2;
      int kq = (lin & 3) << 3;
      async_b128(ldsoff(&sBT[buf][n * 32 + kq]),
                 BT + (size_t)(n0 + n) * K + k0 + kq);
    }
  };

  stage(0, 0);
  wait_async0();
  __syncthreads();
  for (int s = 0; s < nsteps; ++s) {
    int cur = s & 1;
    if (s + 1 < nsteps) stage(s + 1, cur ^ 1);
    wmma_compute(sA[cur], sBT[cur], acc, lane, wm, wn);
    wait_async0();
    __syncthreads();
  }
#pragma unroll
  for (int ns = 0; ns < 4; ++ns) {
    int gn = n0 + wn * 64 + ns * 16 + (lane & 15);
#pragma unroll
    for (int r = 0; r < 8; ++r) {
      int gm = m0 + wm * 16 + ((lane >> 4) << 3) + r;
      C[(size_t)gm * N + gn] = acc[ns][r];
    }
  }
}

// ---------- operand prep ----------
__global__ __launch_bounds__(256) void k_wt_prep(const float* __restrict__ W,  // [1000][800]
                                                 __bf16* __restrict__ WT) {    // [896][1024]
  int idx = blockIdx.x * 256 + threadIdx.x;
  if (idx >= 896 * 1024) return;
  int n = idx >> 10;
  int k = idx & 1023;
  float v = (n < 800 && k < 1000) ? W[(size_t)k * 800 + n] : 0.f;
  WT[idx] = f2bf(v);
}

__global__ __launch_bounds__(256) void k_cvt_bf16(const float* __restrict__ X,
                                                  __bf16* __restrict__ Y, int n) {
  int i = blockIdx.x * 256 + threadIdx.x;
  if (i < n) Y[i] = f2bf(X[i]);
}

// ---------- graph aggregation ----------
__global__ __launch_bounds__(256) void k_zero(float* __restrict__ p, int n) {
  int i = blockIdx.x * 256 + threadIdx.x;
  if (i < n) p[i] = 0.f;
}

__global__ __launch_bounds__(256) void k_edge_scatter(
    const int* __restrict__ src, const int* __restrict__ dst,
    const float* __restrict__ x, int D,
    float* __restrict__ msg, float* __restrict__ cnt) {
  int e = blockIdx.x;
  int s = src[e], d = dst[e];
  const float* xr = x + (size_t)s * D;
  float* mr = msg + (size_t)d * D;
  for (int c = threadIdx.x; c < D; c += 256)
    atomicAdd(&mr[c], xr[c]);
  if (threadIdx.x == 0 && cnt != nullptr)
    atomicAdd(&cnt[d], 1.f);
}

__global__ __launch_bounds__(256) void k_mean_div(float* __restrict__ msg,
                                                  const float* __restrict__ cnt,
                                                  int total, int D) {
  int i = blockIdx.x * 256 + threadIdx.x;
  if (i < total) {
    float c = cnt[i / D];
    msg[i] = msg[i] / fmaxf(c, 1.f);
  }
}

// ---------- LayerNorm ----------
__global__ __launch_bounds__(256) void k_layernorm_f32(
    const float* __restrict__ X, const float* __restrict__ g,
    const float* __restrict__ b, float* __restrict__ Y, int C) {
  int row = blockIdx.x, tid = threadIdx.x;
  __shared__ float s1[256], s2[256];
  float a = 0.f, q = 0.f;
  for (int c = tid; c < C; c += 256) {
    float v = X[(size_t)row * C + c];
    a += v; q += v * v;
  }
  s1[tid] = a; s2[tid] = q;
  __syncthreads();
  for (int o = 128; o > 0; o >>= 1) {
    if (tid < o) { s1[tid] += s1[tid + o]; s2[tid] += s2[tid + o]; }
    __syncthreads();
  }
  float mean = s1[0] / C;
  float var  = s2[0] / C - mean * mean;
  float rstd = rsqrtf(var + 1e-5f);
  for (int c = tid; c < C; c += 256) {
    float v = X[(size_t)row * C + c];
    Y[(size_t)row * C + c] = (v - mean) * rstd * g[c] + b[c];
  }
}

__global__ __launch_bounds__(256) void k_layernorm_bf16_inplace(
    __bf16* __restrict__ X, const float* __restrict__ g,
    const float* __restrict__ b, int C) {
  int row = blockIdx.x, tid = threadIdx.x;
  __shared__ float s1[256], s2[256];
  float a = 0.f, q = 0.f;
  for (int c = tid; c < C; c += 256) {
    float v = bf2f(X[(size_t)row * C + c]);
    a += v; q += v * v;
  }
  s1[tid] = a; s2[tid] = q;
  __syncthreads();
  for (int o = 128; o > 0; o >>= 1) {
    if (tid < o) { s1[tid] += s1[tid + o]; s2[tid] += s2[tid + o]; }
    __syncthreads();
  }
  float mean = s1[0] / C;
  float var  = s2[0] / C - mean * mean;
  float rstd = rsqrtf(var + 1e-5f);
  __syncthreads();
  for (int c = tid; c < C; c += 256) {
    float v = bf2f(X[(size_t)row * C + c]);
    X[(size_t)row * C + c] = f2bf((v - mean) * rstd * g[c] + b[c]);
  }
}

// =======================================================================
extern "C" void kernel_launch(void* const* d_in, const int* in_sizes, int n_in,
                              void* d_out, int out_size, void* d_ws, size_t ws_size,
                              hipStream_t stream) {
  const float* img   = (const float*)d_in[0];
  const float* nodes = (const float*)d_in[1];
  const int*   esrc  = (const int*)d_in[2];
  const int*   edst  = (const int*)d_in[3];
  const float* W1l = (const float*)d_in[4];
  const float* W1r = (const float*)d_in[5];
  const float* b1  = (const float*)d_in[6];
  const float* W2l = (const float*)d_in[7];
  const float* W2r = (const float*)d_in[8];
  const float* b2  = (const float*)d_in[9];
  const float* Wi1 = (const float*)d_in[10];
  const float* bi1 = (const float*)d_in[11];
  const float* Wi2 = (const float*)d_in[12];
  const float* bi2 = (const float*)d_in[13];
  const float* Wi3 = (const float*)d_in[14];
  const float* bi3 = (const float*)d_in[15];
  const float* gi    = (const float*)d_in[16];
  const float* betai = (const float*)d_in[17];
  const float* Wp1 = (const float*)d_in[18];
  const float* bp1 = (const float*)d_in[19];
  const float* Wp2 = (const float*)d_in[20];
  const float* bp2 = (const float*)d_in[21];
  const float* gp    = (const float*)d_in[22];
  const float* betap = (const float*)d_in[23];
  (void)in_sizes; (void)n_in; (void)out_size; (void)ws_size;

  // ---- workspace carve (256B aligned) ----
  char* wsp = (char*)d_ws;
  auto carve = [&](size_t bytes) -> char* {
    char* p = wsp;
    wsp += (bytes + 255) & ~(size_t)255;
    return p;
  };
  float*  msg1 = (float*)carve((size_t)N_NODES * 600 * 4);   // agg1 (in-place mean)
  float*  cnt  = (float*)carve((size_t)N_NODES * 4);
  float*  h    = (float*)carve((size_t)N_NODES * 2048 * 4);
  float*  msg2 = (float*)carve((size_t)N_NODES * 2048 * 4);  // agg2 (in-place mean)
  float*  z    = (float*)carve((size_t)N_NODES * 512 * 4);
  float*  U    = (float*)carve((size_t)NATTRS * 1000 * 4);
  float*  V    = (float*)carve((size_t)NOBJS * 1000 * 4);
  float*  ie1  = (float*)carve((size_t)512 * 800 * 4);
  float*  ie2  = (float*)carve((size_t)512 * 1000 * 4);
  float*  ie3  = (float*)carve((size_t)512 * 800 * 4);
  float*  ieF  = (float*)carve((size_t)512 * 800 * 4);
  __bf16* ieB  = (__bf16*)carve((size_t)512 * 800 * 2);      // ie as bf16 (async A)
  __bf16* WT   = (__bf16*)carve((size_t)896 * 1024 * 2);     // Wp2^T bf16, padded
  __bf16* pe   = (__bf16*)carve((size_t)NPAIRS * 800 * 2);   // 128 MB
  float*  out  = (float*)d_out;                              // [512, 80000]

  // ---- operand prep for async-staged GEMMs ----
  k_wt_prep<<<(896 * 1024) / 256, 256, 0, stream>>>(Wp2, WT);

  // ---- GNN stage 1: agg1 = mean_{j->i} nodes[j]; h = relu(agg1@W1l + nodes@W1r + b1)
  k_zero<<<(N_NODES * 600 + 255) / 256, 256, 0, stream>>>(msg1, N_NODES * 600);
  k_zero<<<(N_NODES + 255) / 256, 256, 0, stream>>>(cnt, N_NODES);
  k_zero<<<(N_NODES * 2048 + 255) / 256, 256, 0, stream>>>(msg2, N_NODES * 2048);
  k_edge_scatter<<<NEDGES, 256, 0, stream>>>(esrc, edst, nodes, 600, msg1, cnt);
  k_mean_div<<<(N_NODES * 600 + 255) / 256, 256, 0, stream>>>(msg1, cnt, N_NODES * 600, 600);
  k_gemm_dual<<<dim3(16, 10), 256, 0, stream>>>(msg1, W1l, nodes, W1r, b1, h,
                                                N_NODES, 2048, 600, 1);
  // ---- GNN stage 2: z = agg2@W2l + h@W2r + b2
  k_edge_scatter<<<NEDGES, 256, 0, stream>>>(esrc, edst, h, 2048, msg2, nullptr);
  k_mean_div<<<(N_NODES * 2048 + 255) / 256, 256, 0, stream>>>(msg2, cnt, N_NODES * 2048, 2048);
  k_gemm_dual<<<dim3(4, 10), 256, 0, stream>>>(msg2, W2l, h, W2r, b2, z,
                                               N_NODES, 512, 2048, 0);
  // ---- pair factorization: U = z[:200]@Wp1_top, V = z[200:]@Wp1_bot
  k_gemm_dual<<<dim3(8, 4), 256, 0, stream>>>(z, Wp1, nullptr, nullptr, nullptr, U,
                                              NATTRS, 1000, 512, 0);
  k_gemm_dual<<<dim3(8, 7), 256, 0, stream>>>(z + (size_t)NATTRS * 512,
                                              Wp1 + (size_t)512 * 1000,
                                              nullptr, nullptr, nullptr, V,
                                              NOBJS, 1000, 512, 0);
  // ---- img branch: 512x512 -> 800 -> 1000 -> 800 -> LN -> bf16
  k_gemm_dual<<<dim3(7, 8), 256, 0, stream>>>(img, Wi1, nullptr, nullptr, bi1, ie1,
                                              512, 800, 512, 1);
  k_gemm_dual<<<dim3(8, 8), 256, 0, stream>>>(ie1, Wi2, nullptr, nullptr, bi2, ie2,
                                              512, 1000, 800, 1);
  k_gemm_dual<<<dim3(7, 8), 256, 0, stream>>>(ie2, Wi3, nullptr, nullptr, bi3, ie3,
                                              512, 800, 1000, 0);
  k_layernorm_f32<<<512, 256, 0, stream>>>(ie3, gi, betai, ieF, 800);
  k_cvt_bf16<<<(512 * 800 + 255) / 256, 256, 0, stream>>>(ieF, ieB, 512 * 800);
  // ---- pair branch: pe = LN(relu(U[i]+V[j]+bp1) @ Wp2 + bp2)  (fused, bf16 out)
  k_gemm_pair<<<dim3(7, NPAIRS / 64), 256, 0, stream>>>(U, V, bp1, WT, bp2, pe);
  k_layernorm_bf16_inplace<<<NPAIRS, 256, 0, stream>>>(pe, gp, betap, 800);
  // ---- final: out = ie @ pe^T   [512, 80000]
  k_gemm_abt<<<dim3(NPAIRS / 128, 8), 256, 0, stream>>>(ieB, pe, out, 512, NPAIRS, 800);
}
